// SwitchFeedForward_3453153706578
// MI455X (gfx1250) — matrix operand order repaired and checked by reading, not compiled, not run on hardware
//
#include <hip/hip_runtime.h>

typedef __attribute__((ext_vector_type(16))) _Float16 v16h;
typedef __attribute__((ext_vector_type(8)))  _Float16 v8h;
typedef __attribute__((ext_vector_type(8)))  float    v8f;
typedef __attribute__((ext_vector_type(4)))  float    v4f;

constexpr int TOK = 8192;   // B*S tokens
constexpr int DD  = 1024;   // model dim
constexpr int FF  = 4096;   // ffn dim
constexpr int EE  = 8;      // experts
constexpr int TM  = 64;     // tokens per block tile (4 WMMA m-tiles)
constexpr int FC  = 512;    // F chunk held in LDS
constexpr int NTHREADS = 512;
constexpr int MAXTILES = (TOK + TM - 1) / TM;   // 128

static __device__ __forceinline__ v8f wmma_f16(v16h a, v16h b, v8f c) {
    return __builtin_amdgcn_wmma_f32_16x16x32_f16(false, a, false, b, (short)0, c,
                                                  false, false);
}

static __device__ __forceinline__ v8f v8f_zero() {
    v8f z = {0.f, 0.f, 0.f, 0.f, 0.f, 0.f, 0.f, 0.f};
    return z;
}

// tanh-approx gelu (jax.nn.gelu approximate=True), sigmoid form:
// 0.5x(1+tanh(z)) == x * sigmoid(2z). v_exp_f32 + v_rcp_f32, no IEEE divide.
static __device__ __forceinline__ float gelu_tanh(float x) {
    float z2 = 1.5957691216057308f * (x + 0.044715f * x * x * x);
    return x * __builtin_amdgcn_rcpf(1.f + __expf(-z2));
}

// A fragment (16x32 f16) from LDS row-major matrix.
// ISA layout: lanes 0-15 row M=lane, K 0..7 & 16..23; lanes 16-31 row M=lane-16, K 8..15 & 24..31.
static __device__ __forceinline__ v16h load_a_lds(const _Float16* mat, int ld,
                                                  int rowBase, int kBase, int lane) {
    int r  = rowBase + (lane & 15);
    int hs = lane >> 4;
    const _Float16* p = mat + (size_t)r * ld + kBase + hs * 8;
    v8h lo = *(const v8h*)(p);
    v8h hi = *(const v8h*)(p + 16);
    return __builtin_shufflevector(lo, hi, 0,1,2,3,4,5,6,7,8,9,10,11,12,13,14,15);
}

// B fragment (32x16) from global row-major fp32 matrix (row stride ld), convert to f16.
// ISA layout: lanes 0-15 col N=lane, K 0..7 & 16..23; lanes 16-31 same cols, K 8..15 & 24..31.
static __device__ __forceinline__ v16h load_b_f32(const float* mat, long ld,
                                                  long kBase, long col, int lane) {
    long k0 = kBase + (long)(lane >> 4) * 8;
    const float* p = mat + k0 * ld + col;
    v16h b;
#pragma unroll
    for (int i = 0; i < 16; ++i) b[i] = (_Float16)p[(long)(i < 8 ? i : i + 8) * ld];
    return b;
}

// One GEMM1 n-tile pass: acc[4] += Xs(64xDD) @ W1[:, col..col+16)
static __device__ __forceinline__ void gemm1_pass(const _Float16* Xs, const float* w1e,
                                                  long col, int lane, v8f acc[4]) {
    for (int kb = 0; kb < DD / 32; ++kb) {
        const int kBase = kb * 32;
        v16h a[4];
#pragma unroll
        for (int mt = 0; mt < 4; ++mt)
            a[mt] = load_a_lds(Xs, DD, mt * 16, kBase, lane);
        v16h b = load_b_f32(w1e, FF, kBase, col, lane);
#pragma unroll
        for (int mt = 0; mt < 4; ++mt)
            acc[mt] = wmma_f16(a[mt], b, acc[mt]);
    }
}

// bias + gelu + f16 store of one n-tile's accumulators into Hs
static __device__ __forceinline__ void hs_write_pass(_Float16* Hs, const v8f acc[4],
                                                     float bb, int colL, int hs) {
#pragma unroll
    for (int mt = 0; mt < 4; ++mt) {
#pragma unroll
        for (int j = 0; j < 8; ++j) {
            int r = mt * 16 + hs * 8 + j;
            Hs[(size_t)r * FC + colL] = (_Float16)gelu_tanh(acc[mt][j] + bb);
        }
    }
}

// ---------------- router: logits -> softmax max prob + argmax ----------------
__global__ __launch_bounds__(256) void moe_router(const float* __restrict__ x,
                                                  const float* __restrict__ sw,
                                                  const float* __restrict__ sb,
                                                  int* __restrict__ routes,
                                                  float* __restrict__ pmax) {
    int wave = threadIdx.x >> 5;
    int lane = threadIdx.x & 31;
    int t = blockIdx.x * 8 + wave;
    if (t >= TOK) return;

    float acc[EE];
#pragma unroll
    for (int e = 0; e < EE; ++e) acc[e] = 0.f;

    const float* xr = x + (size_t)t * DD;
    for (int i = 0; i < DD / 32; ++i) {
        int d = i * 32 + lane;
        float xv = xr[d];
        const float4* swr = (const float4*)(sw + (size_t)d * EE);
        float4 a = swr[0], b = swr[1];
        acc[0] += xv * a.x; acc[1] += xv * a.y; acc[2] += xv * a.z; acc[3] += xv * a.w;
        acc[4] += xv * b.x; acc[5] += xv * b.y; acc[6] += xv * b.z; acc[7] += xv * b.w;
    }
#pragma unroll
    for (int e = 0; e < EE; ++e) {
#pragma unroll
        for (int off = 16; off >= 1; off >>= 1)
            acc[e] += __shfl_xor(acc[e], off, 32);
    }
    if (lane == 0) {
        float lg[EE];
        float m = -1e30f;
        int arg = 0;
#pragma unroll
        for (int e = 0; e < EE; ++e) {
            lg[e] = acc[e] + sb[e];
            if (lg[e] > m) { m = lg[e]; arg = e; }   // first-max, matches argmax
        }
        float s = 0.f;
#pragma unroll
        for (int e = 0; e < EE; ++e) s += __expf(lg[e] - m);
        routes[t] = arg;
        pmax[t]   = 1.f / s;   // exp(m-m)/sum == max softmax prob
    }
}

__global__ void moe_zero_counts(int* counts) {
    if (threadIdx.x < EE) counts[threadIdx.x] = 0;
}

__global__ __launch_bounds__(256) void moe_scatter(const int* __restrict__ routes,
                                                   int* __restrict__ counts,
                                                   int* __restrict__ idx) {
    int t = blockIdx.x * blockDim.x + threadIdx.x;
    if (t >= TOK) return;
    int e = routes[t];
    int pos = atomicAdd(&counts[e], 1);
    idx[e * TOK + pos] = t;
}

// ---------------- fused expert FFN: Y = gelu(X @ W1 + b1) @ W2 + b2, scaled ----------------
__global__ __launch_bounds__(NTHREADS) void moe_ffn(const float* __restrict__ x,
                                                    const float* __restrict__ w1,
                                                    const float* __restrict__ b1,
                                                    const float* __restrict__ w2,
                                                    const float* __restrict__ b2,
                                                    const int* __restrict__ idx,
                                                    const int* __restrict__ counts,
                                                    const float* __restrict__ pmax,
                                                    float* __restrict__ out) {
    const int e    = blockIdx.x / MAXTILES;
    const int tile = blockIdx.x % MAXTILES;
    const int cnt  = counts[e];
    const int tileBase = tile * TM;
    if (tileBase >= cnt) return;

    extern __shared__ char smem[];
    _Float16* Xs   = (_Float16*)smem;                         // TM x DD   (128 KB)
    _Float16* Hs   = (_Float16*)(smem + TM * DD * 2);         // TM x FC   (64 KB)
    int*      toks = (int*)(smem + TM * DD * 2 + TM * FC * 2);
    float*    scl  = (float*)(toks + TM);

    const int tid  = threadIdx.x;
    const int wave = tid >> 5;
    const int lane = tid & 31;
    const int lane15 = lane & 15;
    const int hs = lane >> 4;

    const float* w1e = w1 + (size_t)e * DD * FF;
    const float* b1e = b1 + (size_t)e * FF;
    const float* w2e = w2 + (size_t)e * FF * DD;
    const float* b2e = b2 + (size_t)e * DD;

    // token indices + per-token scale for this tile
    if (tid < TM) {
        int pos = tileBase + tid;
        int tk = (pos < cnt) ? idx[e * TOK + pos] : -1;
        toks[tid] = tk;
        scl[tid]  = (tk >= 0) ? pmax[tk] : 0.f;
    }
    __syncthreads();

    // stage X tile (fp32 -> f16) into LDS; padded rows are zeros.
    // Non-temporal: x is streamed once, keep L2 for expert weights.
    for (int i = tid; i < TM * (DD / 4); i += NTHREADS) {
        int r  = i / (DD / 4);
        int c4 = i % (DD / 4);
        int tk = toks[r];
        v4f v = {0.f, 0.f, 0.f, 0.f};
        if (tk >= 0)
            v = __builtin_nontemporal_load((const v4f*)(x + (size_t)tk * DD) + c4);
        _Float16* dst = Xs + (size_t)r * DD + c4 * 4;
        dst[0] = (_Float16)v.x; dst[1] = (_Float16)v.y;
        dst[2] = (_Float16)v.z; dst[3] = (_Float16)v.w;
    }
    __syncthreads();

    // persistent Y accumulators: 4 m-tiles x 4 n-tiles of 16 cols -> 64 D-cols/wave
    v8f acc2[4][4];
#pragma unroll
    for (int mt = 0; mt < 4; ++mt)
#pragma unroll
        for (int nt = 0; nt < 4; ++nt) acc2[mt][nt] = v8f_zero();

    const int g1col0 = wave * 32;   // this wave's F columns within chunk (2 n-tiles)
    const int g2col0 = wave * 64;   // this wave's D columns (4 n-tiles)

    for (int fc = 0; fc < FF / FC; ++fc) {
        const long fcBase = (long)fc * FC;

        // ---- GEMM1 n-tile 0 (computed before the barrier to overlap GEMM2 drain) ----
        v8f acc1[4];
#pragma unroll
        for (int mt = 0; mt < 4; ++mt) acc1[mt] = v8f_zero();
        gemm1_pass(Xs, w1e, fcBase + g1col0 + lane15, lane, acc1);

        __syncthreads();   // previous chunk's Hs fully consumed by all waves

        {   // write n-tile 0
            int colL = g1col0 + lane15;
            hs_write_pass(Hs, acc1, b1e[fcBase + colL], colL, hs);
        }

        // ---- GEMM1 n-tile 1 ----
#pragma unroll
        for (int mt = 0; mt < 4; ++mt) acc1[mt] = v8f_zero();
        gemm1_pass(Xs, w1e, fcBase + g1col0 + 16 + lane15, lane, acc1);
        {   // write n-tile 1
            int colL = g1col0 + 16 + lane15;
            hs_write_pass(Hs, acc1, b1e[fcBase + colL], colL, hs);
        }
        __syncthreads();

        // ---- GEMM2 partial: acc2 += Hs @ W2[chunk rows, :] ----
        for (int kb = 0; kb < FC / 32; ++kb) {
            const int kBase = kb * 32;
            v16h a[4];
#pragma unroll
            for (int mt = 0; mt < 4; ++mt)
                a[mt] = load_a_lds(Hs, FC, mt * 16, kBase, lane);
#pragma unroll
            for (int nt = 0; nt < 4; ++nt) {
                long col = g2col0 + nt * 16 + lane15;
                v16h b = load_b_f32(w2e, DD, fcBase + kBase, col, lane);
#pragma unroll
                for (int mt = 0; mt < 4; ++mt)
                    acc2[mt][nt] = wmma_f16(a[mt], b, acc2[mt][nt]);
            }
        }
    }

    // epilogue: (y + b2) * route_prob_max, scattered back to token rows.
    // Non-temporal stores: out is write-once, keep L2 for expert weights.
#pragma unroll
    for (int nt = 0; nt < 4; ++nt) {
        int colL = g2col0 + nt * 16 + lane15;
        float bb = b2e[colL];
#pragma unroll
        for (int mt = 0; mt < 4; ++mt) {
#pragma unroll
            for (int j = 0; j < 8; ++j) {
                int r = mt * 16 + hs * 8 + j;
                int tk = toks[r];
                if (tk >= 0)
                    __builtin_nontemporal_store((acc2[mt][nt][j] + bb) * scl[r],
                                                out + (size_t)tk * DD + colL);
            }
        }
    }
}

extern "C" void kernel_launch(void* const* d_in, const int* in_sizes, int n_in,
                              void* d_out, int out_size, void* d_ws, size_t ws_size,
                              hipStream_t stream) {
    const float* x   = (const float*)d_in[0];
    const float* sw  = (const float*)d_in[1];
    const float* sb  = (const float*)d_in[2];
    const float* w1  = (const float*)d_in[3];
    const float* b1  = (const float*)d_in[4];
    const float* w2  = (const float*)d_in[5];
    const float* b2  = (const float*)d_in[6];
    float* out = (float*)d_out;

    // workspace layout (ints/floats): routes[T] | pmax[T] | counts[E] | idx[E*T]
    int*   routes = (int*)d_ws;
    float* pmax   = (float*)(routes + TOK);
    int*   counts = (int*)(pmax + TOK);
    int*   idx    = counts + EE;

    moe_zero_counts<<<1, 32, 0, stream>>>(counts);
    moe_router<<<TOK / 8, 256, 0, stream>>>(x, sw, sb, routes, pmax);
    moe_scatter<<<TOK / 256, 256, 0, stream>>>(routes, counts, idx);

    const size_t smem = (size_t)TM * DD * 2 + (size_t)TM * FC * 2
                      + (size_t)TM * (sizeof(int) + sizeof(float));   // ~197 KB < 320 KB
    moe_ffn<<<EE * MAXTILES, NTHREADS, smem, stream>>>(x, w1, b1, w2, b2,
                                                       idx, counts, pmax, out);
}
